// ECLIPSE_51917564674587
// MI455X (gfx1250) — matrix operand order repaired and checked by reading, not compiled, not run on hardware
//
#include <hip/hip_runtime.h>
#include <hip/hip_bf16.h>
#include <math.h>

#ifndef __has_builtin
#define __has_builtin(x) 0
#endif

#define H_DIM 128
#define HEADS 8
#define DHEAD 16
#define N_CMP 40000
#define N_PRT 15000
#define F_CMP 1024
#define F_PRT 1280
#define E_CP 200000
#define E_PC 200000
#define E_PP 100000
#define E_CC 100000

typedef __attribute__((ext_vector_type(16))) __bf16 v16bf;
typedef __attribute__((ext_vector_type(8)))  float v8f;
typedef __attribute__((ext_vector_type(2)))  float v2f;

union ABfrag { unsigned u[8]; v16bf v; };

__device__ __forceinline__ unsigned short f2bf(float f) {
  unsigned u = __float_as_uint(f);
  u += 0x7fffu + ((u >> 16) & 1u);          // round-to-nearest-even
  return (unsigned short)(u >> 16);
}
__device__ __forceinline__ unsigned packbf2(float lo, float hi) {
  return (unsigned)f2bf(lo) | ((unsigned)f2bf(hi) << 16);
}

enum { EPI_NONE = 0, EPI_RELU = 1, EPI_SKIP = 2 };

// OUT[M,128] = epi( A[M,K] @ W[K,128] + bias ), bf16 WMMA, fp32 accumulate.
// Block = 256 threads (8 waves), 128x128 output tile; wave w owns rows [16w,16w+16).
template <int EPI>
__global__ __launch_bounds__(256)
void gemm128(const float* __restrict__ A, int M, int K,
             const float* __restrict__ W, const float* __restrict__ bias,
             const float* __restrict__ Hprev, const float* __restrict__ skipgate,
             float* __restrict__ OUT) {
  // As[m][kk] packs bf16(A[m][2kk]) | bf16(A[m][2kk+1])<<16  (kk = K/2 within 32-chunk)
  __shared__ unsigned As[128][17];
  // Ws[kk][n] packs bf16(W[2kk][n]) | bf16(W[2kk+1][n])<<16
  __shared__ unsigned Ws[16][132];

  const int tid  = threadIdx.x;
  const int wave = tid >> 5;
  const int lane = tid & 31;
  const int half = lane >> 4;
  const int l15  = lane & 15;
  const int m0   = blockIdx.x * 128;

  const v8f vzero = {0.f,0.f,0.f,0.f,0.f,0.f,0.f,0.f};
  v8f acc[8];
#pragma unroll
  for (int c = 0; c < 8; ++c) acc[c] = vzero;

  for (int k0 = 0; k0 < K; k0 += 32) {
    // ---- load + convert A tile: 128 rows x 32 cols ----
    {
      int m   = tid >> 1;
      int kk0 = (tid & 1) * 8;
      int gm  = m0 + m;
      int gmc = gm < M ? gm : M - 1;
      const float* src = A + (size_t)gmc * K + k0 + kk0 * 2;
      float4 f0 = ((const float4*)src)[0];
      float4 f1 = ((const float4*)src)[1];
      float4 f2 = ((const float4*)src)[2];
      float4 f3 = ((const float4*)src)[3];
      As[m][kk0+0] = packbf2(f0.x, f0.y);
      As[m][kk0+1] = packbf2(f0.z, f0.w);
      As[m][kk0+2] = packbf2(f1.x, f1.y);
      As[m][kk0+3] = packbf2(f1.z, f1.w);
      As[m][kk0+4] = packbf2(f2.x, f2.y);
      As[m][kk0+5] = packbf2(f2.z, f2.w);
      As[m][kk0+6] = packbf2(f3.x, f3.y);
      As[m][kk0+7] = packbf2(f3.z, f3.w);
    }
    // ---- load + convert W tile: 32 rows x 128 cols, packed K-pairwise ----
    {
      int kk = tid >> 4;                 // 0..15
      int n0 = (tid & 15) * 8;
      const float* w0 = W + (size_t)(k0 + 2 * kk) * H_DIM + n0;
      float4 e0 = ((const float4*)w0)[0];
      float4 e1 = ((const float4*)w0)[1];
      float4 o0 = ((const float4*)(w0 + H_DIM))[0];
      float4 o1 = ((const float4*)(w0 + H_DIM))[1];
      Ws[kk][n0+0] = packbf2(e0.x, o0.x);
      Ws[kk][n0+1] = packbf2(e0.y, o0.y);
      Ws[kk][n0+2] = packbf2(e0.z, o0.z);
      Ws[kk][n0+3] = packbf2(e0.w, o0.w);
      Ws[kk][n0+4] = packbf2(e1.x, o1.x);
      Ws[kk][n0+5] = packbf2(e1.y, o1.y);
      Ws[kk][n0+6] = packbf2(e1.z, o1.z);
      Ws[kk][n0+7] = packbf2(e1.w, o1.w);
    }
    __syncthreads();

    // A fragment (16x32 bf16): lanes0-15 K=0..7,16..23 ; lanes16-31 K=8..15,24..31
    ABfrag a;
    const int am = wave * 16 + l15;
#pragma unroll
    for (int j = 0; j < 8; ++j) {
      int kk = (j < 4 ? j : j + 4) + half * 4;
      a.u[j] = As[am][kk];
    }
#pragma unroll
    for (int c = 0; c < 8; ++c) {
      // B fragment (32x16 bf16): lanes0-15 K=0..7,16..23 ; lanes16-31 K=8..15,24..31
      ABfrag b;
#pragma unroll
      for (int j = 0; j < 8; ++j) b.u[j] = Ws[(j < 4 ? j : j + 4) + half * 4][c * 16 + l15];
      acc[c] = __builtin_amdgcn_wmma_f32_16x16x32_bf16(
          false, a.v, false, b.v, (short)0, acc[c], false, false);
    }
    __syncthreads();
  }

  float sg = 0.f;
  if (EPI == EPI_SKIP) sg = 1.f / (1.f + expf(-skipgate[0]));
  const int mbase = m0 + wave * 16 + half * 8;   // C/D layout: vgpr r -> row mbase+r
  if (m0 + 128 <= M) {
    // fast path: whole tile in range, straight-line coalesced stores
#pragma unroll
    for (int c = 0; c < 8; ++c) {
      int n = c * 16 + l15;
      float bn = bias[n];
#pragma unroll
      for (int r = 0; r < 8; ++r) {
        float val = acc[c][r] + bn;
        if (EPI == EPI_RELU) val = val > 0.f ? val : 0.f;
        if (EPI == EPI_SKIP)
          val = sg * val + (1.f - sg) * Hprev[(size_t)(mbase + r) * H_DIM + n];
        OUT[(size_t)(mbase + r) * H_DIM + n] = val;
      }
    }
  } else {
#pragma unroll
    for (int c = 0; c < 8; ++c) {
      int n = c * 16 + l15;
      float bn = bias[n];
#pragma unroll
      for (int r = 0; r < 8; ++r) {
        int m = mbase + r;
        if (m < M) {
          float val = acc[c][r] + bn;
          if (EPI == EPI_RELU) val = val > 0.f ? val : 0.f;
          if (EPI == EPI_SKIP)
            val = sg * val + (1.f - sg) * Hprev[(size_t)m * H_DIM + n];
          OUT[(size_t)m * H_DIM + n] = val;
        }
      }
    }
  }
}

// Per-head relation transforms: krel[n,h,:] = K[n,h,:] @ a_rel[h]; vrel likewise.
// One wave per (16-node tile, head). fp32 WMMA 16x16x4, K-loop of 4.
__global__ __launch_bounds__(256)
void rel_transform(const float* __restrict__ Kmat, const float* __restrict__ Vmat,
                   const float* __restrict__ a_rel, const float* __restrict__ m_rel,
                   float* __restrict__ krel, float* __restrict__ vrel, int N) {
  const int h    = threadIdx.x >> 5;
  const int lane = threadIdx.x & 31;
  const int half = lane >> 4, l15 = lane & 15;
  const int n0   = blockIdx.x * 16;
  int rn = n0 + l15;
  int rc = rn < N ? rn : N - 1;
  const float* krow = Kmat + (size_t)rc * H_DIM + h * DHEAD;
  const float* vrow = Vmat + (size_t)rc * H_DIM + h * DHEAD;
  const float* arl  = a_rel + h * DHEAD * DHEAD;
  const float* mrl  = m_rel + h * DHEAD * DHEAD;

#if __has_builtin(__builtin_amdgcn_wmma_f32_16x16x4_f32)
  const v8f vzero = {0.f,0.f,0.f,0.f,0.f,0.f,0.f,0.f};
  v8f ka = vzero, va = vzero;
#pragma unroll
  for (int s = 0; s < 4; ++s) {
    // A 16x4 f32: lanes0-15 {K0,K1}, lanes16-31 {K2,K3} (per 4-step)
    int kidx = 4 * s + half * 2;
    v2f ak; ak.x = krow[kidx]; ak.y = krow[kidx + 1];
    v2f av; av.x = vrow[kidx]; av.y = vrow[kidx + 1];
    v2f bk; bk.x = arl[kidx * DHEAD + l15]; bk.y = arl[(kidx + 1) * DHEAD + l15];
    v2f bv; bv.x = mrl[kidx * DHEAD + l15]; bv.y = mrl[(kidx + 1) * DHEAD + l15];
    ka = __builtin_amdgcn_wmma_f32_16x16x4_f32(false, ak, false, bk, (short)0, ka, false, false);
    va = __builtin_amdgcn_wmma_f32_16x16x4_f32(false, av, false, bv, (short)0, va, false, false);
  }
  if (n0 + 16 <= N) {
    float* kdst = krel + (size_t)(n0 + half * 8) * H_DIM + h * DHEAD + l15;
    float* vdst = vrel + (size_t)(n0 + half * 8) * H_DIM + h * DHEAD + l15;
#pragma unroll
    for (int r = 0; r < 8; ++r) {
      kdst[(size_t)r * H_DIM] = ka[r];
      vdst[(size_t)r * H_DIM] = va[r];
    }
  } else {
#pragma unroll
    for (int r = 0; r < 8; ++r) {
      int m = n0 + half * 8 + r;
      if (m < N) {
        krel[(size_t)m * H_DIM + h * DHEAD + l15] = ka[r];
        vrel[(size_t)m * H_DIM + h * DHEAD + l15] = va[r];
      }
    }
  }
#else
  // VALU fallback
  for (int r = 0; r < 8; ++r) {
    int m = n0 + half * 8 + r;
    if (m >= N) continue;
    const float* kr = Kmat + (size_t)m * H_DIM + h * DHEAD;
    const float* vr = Vmat + (size_t)m * H_DIM + h * DHEAD;
    float sk = 0.f, sv = 0.f;
    for (int dd = 0; dd < DHEAD; ++dd) {
      sk += kr[dd] * arl[dd * DHEAD + l15];
      sv += vr[dd] * mrl[dd * DHEAD + l15];
    }
    krel[(size_t)m * H_DIM + h * DHEAD + l15] = sk;
    vrel[(size_t)m * H_DIM + h * DHEAD + l15] = sv;
  }
#endif
}

// Pass A: per (edge,head) attention logit + order-preserving atomic max per dst node.
__global__ void edge_scores(const int* __restrict__ src, const int* __restrict__ dst,
                            const float* __restrict__ q, const float* __restrict__ krel,
                            const float* __restrict__ prel,
                            float* __restrict__ abuf, unsigned* __restrict__ maxenc, int E) {
  int idx = blockIdx.x * blockDim.x + threadIdx.x;
  if (idx >= E * HEADS) return;
  int e = idx >> 3, h = idx & 7;
  int s = src[e], d = dst[e];
  const float4* qp = (const float4*)(q    + (size_t)d * H_DIM + h * DHEAD);
  const float4* kp = (const float4*)(krel + (size_t)s * H_DIM + h * DHEAD);
  float acc = 0.f;
#pragma unroll
  for (int i = 0; i < 4; ++i) {
    float4 qa = qp[i], kb = kp[i];
    acc += qa.x * kb.x + qa.y * kb.y + qa.z * kb.z + qa.w * kb.w;
  }
  float a = acc * prel[h] * 0.25f;     // / sqrt(D=16)
  abuf[idx] = a;
  unsigned bits = __float_as_uint(a);
  unsigned key  = (bits & 0x80000000u) ? ~bits : (bits | 0x80000000u);
  atomicMax(maxenc + (size_t)d * HEADS + h, key);
}

// Pass B: ex = exp(a - max); den[dst] += ex; agg[dst] += ex * vrel[src].
__global__ void edge_accum(const int* __restrict__ src, const int* __restrict__ dst,
                           const float* __restrict__ abuf, const unsigned* __restrict__ maxenc,
                           float* __restrict__ den, const float* __restrict__ vrel,
                           float* __restrict__ agg, int E) {
  int idx = blockIdx.x * blockDim.x + threadIdx.x;
  if (idx >= E * HEADS) return;
  int e = idx >> 3, h = idx & 7;
  int s = src[e], d = dst[e];
  unsigned key = maxenc[(size_t)d * HEADS + h];
  float mx = (key & 0x80000000u) ? __uint_as_float(key & 0x7fffffffu)
                                 : __uint_as_float(~key);
  float ex = expf(abuf[idx] - mx);
  unsafeAtomicAdd(den + (size_t)d * HEADS + h, ex);
  const float4* vp = (const float4*)(vrel + (size_t)s * H_DIM + h * DHEAD);
  float* ag = agg + (size_t)d * H_DIM + h * DHEAD;
#pragma unroll
  for (int i = 0; i < 4; ++i) {
    float4 v = vp[i];
    unsafeAtomicAdd(ag + 4 * i + 0, ex * v.x);
    unsafeAtomicAdd(ag + 4 * i + 1, ex * v.y);
    unsafeAtomicAdd(ag + 4 * i + 2, ex * v.z);
    unsafeAtomicAdd(ag + 4 * i + 3, ex * v.w);
  }
}

// agg <- gelu(agg / (den + 1e-16)), exact gelu
__global__ void gelu_norm(float* __restrict__ agg, const float* __restrict__ den, int N) {
  int idx = blockIdx.x * blockDim.x + threadIdx.x;
  if (idx >= N * H_DIM) return;
  int n = idx >> 7;
  int h = (idx >> 4) & 7;
  float t = agg[idx] / (den[(size_t)n * HEADS + h] + 1e-16f);
  agg[idx] = 0.5f * t * (1.f + erff(t * 0.70710678118654752f));
}

__global__ void zero32(unsigned* __restrict__ p, long n) {
  long i = (long)blockIdx.x * blockDim.x + threadIdx.x;
  if (i < n) p[i] = 0u;
}

// out[i] = dot(cmp_lin[row[i]], prt_lin[col[i]]) / sqrt(128)
__global__ void link_score(const int* __restrict__ row, const int* __restrict__ col,
                           const float* __restrict__ cl, const float* __restrict__ pl,
                           float* __restrict__ out, int NL) {
  int i = blockIdx.x * blockDim.x + threadIdx.x;
  if (i >= NL) return;
  int r = row[i], c = col[i];
  const float4* a = (const float4*)(cl + (size_t)r * H_DIM);
  const float4* b = (const float4*)(pl + (size_t)c * H_DIM);
  float acc = 0.f;
#pragma unroll
  for (int j = 0; j < 32; ++j) {
    float4 x = a[j], y = b[j];
    acc += x.x * y.x + x.y * y.y + x.z * y.z + x.w * y.w;
  }
  out[i] = acc * 0.08838834764831845f;   // 1/sqrt(128)
}

extern "C" void kernel_launch(void* const* d_in, const int* in_sizes, int n_in,
                              void* d_out, int out_size, void* d_ws, size_t ws_size,
                              hipStream_t stream) {
  (void)n_in; (void)out_size; (void)ws_size;
  // Input order (jax pytree flatten, dict keys sorted):
  //  0 x_cmp, 1 x_prt, 2 label_row, 3 label_col,
  //  4 src_cp, 5 dst_cp, 6 src_pc, 7 dst_pc, 8 src_pp, 9 dst_pp, 10 src_cc, 11 dst_cc,
  //  12.. params: convs[l]{ edge{cc,cp,pc,pp}x{a_rel,m_rel,p_rel},
  //                         node{cmp,prt}x{a.b,a.w,k.b,k.w,q.b,q.w,skip,v.b,v.w} },
  //  then left.b,left.w, lin_cmp.b,lin_cmp.w, lin_prt.b,lin_prt.w, right.b,right.w
  const float* x_cmp = (const float*)d_in[0];
  const float* x_prt = (const float*)d_in[1];
  const int* lrow = (const int*)d_in[2];
  const int* lcol = (const int*)d_in[3];
  auto P = [&](int i) { return (const float*)d_in[i]; };

  float* ws = (float*)d_ws;
  size_t off = 0;
  auto alloc = [&](size_t n) { float* p = ws + off; off += n; return p; };
  const size_t SC = (size_t)N_CMP * H_DIM, SP = (size_t)N_PRT * H_DIM;

  float* h_cmp  = alloc(SC); float* h_prt  = alloc(SP);
  float* hn_cmp = alloc(SC); float* hn_prt = alloc(SP);
  float* k_cmp = alloc(SC); float* q_cmp = alloc(SC); float* v_cmp = alloc(SC);
  float* k_prt = alloc(SP); float* q_prt = alloc(SP); float* v_prt = alloc(SP);
  float* krel_s  = alloc(SC);                 // shared scratch (max of both types)
  float* vrel_cp = alloc(SC); float* vrel_cc = alloc(SC);
  float* vrel_pc = alloc(SP); float* vrel_pp = alloc(SP);
  float* a_cp = alloc((size_t)E_CP * 8); float* a_pc = alloc((size_t)E_PC * 8);
  float* a_pp = alloc((size_t)E_PP * 8); float* a_cc = alloc((size_t)E_CC * 8);
  float* stat_cmp = alloc((size_t)N_CMP * (16 + H_DIM));   // [max | den | agg]
  unsigned* max_cmp = (unsigned*)stat_cmp;
  float* den_cmp = stat_cmp + (size_t)N_CMP * 8;
  float* agg_cmp = stat_cmp + (size_t)N_CMP * 16;
  float* stat_prt = alloc((size_t)N_PRT * (16 + H_DIM));
  unsigned* max_prt = (unsigned*)stat_prt;
  float* den_prt = stat_prt + (size_t)N_PRT * 8;
  float* agg_prt = stat_prt + (size_t)N_PRT * 16;

  dim3 blk(256);
  const int gcmp = (N_CMP + 127) / 128, gprt = (N_PRT + 127) / 128;

  // input projections + relu
  gemm128<EPI_RELU><<<gcmp, blk, 0, stream>>>(x_cmp, N_CMP, F_CMP, P(75), P(74), nullptr, nullptr, h_cmp);
  gemm128<EPI_RELU><<<gprt, blk, 0, stream>>>(x_prt, N_PRT, F_PRT, P(77), P(76), nullptr, nullptr, h_prt);

  float* hc = h_cmp; float* hp = h_prt; float* hcn = hn_cmp; float* hpn = hn_prt;

  for (int l = 0; l < 2; ++l) {
    const int B = 12 + 30 * l;
    // K/Q/V projections
    gemm128<EPI_NONE><<<gcmp, blk, 0, stream>>>(hc, N_CMP, H_DIM, P(B+15), P(B+14), nullptr, nullptr, k_cmp);
    gemm128<EPI_NONE><<<gcmp, blk, 0, stream>>>(hc, N_CMP, H_DIM, P(B+17), P(B+16), nullptr, nullptr, q_cmp);
    gemm128<EPI_NONE><<<gcmp, blk, 0, stream>>>(hc, N_CMP, H_DIM, P(B+20), P(B+19), nullptr, nullptr, v_cmp);
    gemm128<EPI_NONE><<<gprt, blk, 0, stream>>>(hp, N_PRT, H_DIM, P(B+24), P(B+23), nullptr, nullptr, k_prt);
    gemm128<EPI_NONE><<<gprt, blk, 0, stream>>>(hp, N_PRT, H_DIM, P(B+26), P(B+25), nullptr, nullptr, q_prt);
    gemm128<EPI_NONE><<<gprt, blk, 0, stream>>>(hp, N_PRT, H_DIM, P(B+29), P(B+28), nullptr, nullptr, v_prt);

    // zero max/den/agg
    long zc = (long)N_CMP * (16 + H_DIM);
    long zp = (long)N_PRT * (16 + H_DIM);
    zero32<<<(int)((zc + 255) / 256), blk, 0, stream>>>((unsigned*)stat_cmp, zc);
    zero32<<<(int)((zp + 255) / 256), blk, 0, stream>>>((unsigned*)stat_prt, zp);

    // edges in order cp, pc, pp, cc (param order within layer: cc=B+0, cp=B+3, pc=B+6, pp=B+9)
    const int* esrc[4]  = {(const int*)d_in[4], (const int*)d_in[6], (const int*)d_in[8], (const int*)d_in[10]};
    const int* edst[4]  = {(const int*)d_in[5], (const int*)d_in[7], (const int*)d_in[9], (const int*)d_in[11]};
    const int  esz[4]   = {E_CP, E_PC, E_PP, E_CC};
    const int  erel[4]  = {B + 3, B + 6, B + 9, B + 0};
    const float* eks[4] = {k_cmp, k_prt, k_prt, k_cmp};
    const float* evs[4] = {v_cmp, v_prt, v_prt, v_cmp};
    const int  ensrc[4] = {N_CMP, N_PRT, N_PRT, N_CMP};
    const float* eq[4]  = {q_prt, q_cmp, q_prt, q_cmp};
    float* evrel[4]     = {vrel_cp, vrel_pc, vrel_pp, vrel_cc};
    float* eabuf[4]     = {a_cp, a_pc, a_pp, a_cc};
    unsigned* emax[4]   = {max_prt, max_cmp, max_prt, max_cmp};
    float* eden[4]      = {den_prt, den_cmp, den_prt, den_cmp};
    float* eagg[4]      = {agg_prt, agg_cmp, agg_prt, agg_cmp};

    for (int i = 0; i < 4; ++i) {
      int grel = (ensrc[i] + 15) / 16;
      rel_transform<<<grel, blk, 0, stream>>>(eks[i], evs[i], P(erel[i]), P(erel[i] + 1),
                                              krel_s, evrel[i], ensrc[i]);
      int gt = (esz[i] * 8 + 255) / 256;
      edge_scores<<<gt, blk, 0, stream>>>(esrc[i], edst[i], eq[i], krel_s, P(erel[i] + 2),
                                          eabuf[i], emax[i], esz[i]);
    }
    for (int i = 0; i < 4; ++i) {
      int gt = (esz[i] * 8 + 255) / 256;
      edge_accum<<<gt, blk, 0, stream>>>(esrc[i], edst[i], eabuf[i], emax[i], eden[i],
                                         evrel[i], eagg[i], esz[i]);
    }

    gelu_norm<<<(N_CMP * H_DIM + 255) / 256, blk, 0, stream>>>(agg_cmp, den_cmp, N_CMP);
    gelu_norm<<<(N_PRT * H_DIM + 255) / 256, blk, 0, stream>>>(agg_prt, den_prt, N_PRT);

    // output projection with sigmoid skip blend
    gemm128<EPI_SKIP><<<gcmp, blk, 0, stream>>>(agg_cmp, N_CMP, H_DIM, P(B+13), P(B+12), hc, P(B+18), hcn);
    gemm128<EPI_SKIP><<<gprt, blk, 0, stream>>>(agg_prt, N_PRT, H_DIM, P(B+22), P(B+21), hp, P(B+27), hpn);

    float* t;
    t = hc; hc = hcn; hcn = t;
    t = hp; hp = hpn; hpn = t;
  }

  // final left/right projections (reuse k_* buffers) + gathered dot scoring
  gemm128<EPI_NONE><<<gcmp, blk, 0, stream>>>(hc, N_CMP, H_DIM, P(73), P(72), nullptr, nullptr, k_cmp);
  gemm128<EPI_NONE><<<gprt, blk, 0, stream>>>(hp, N_PRT, H_DIM, P(79), P(78), nullptr, nullptr, k_prt);

  int NL = in_sizes[2];
  link_score<<<(NL + 255) / 256, blk, 0, stream>>>(lrow, lcol, k_cmp, k_prt, (float*)d_out, NL);
}
